// GATModel_6279242187129
// MI455X (gfx1250) — matrix operand order repaired and checked by reading, not compiled, not run on hardware
//
#include <hip/hip_runtime.h>
#include <hip/hip_bf16.h>
#include <stdint.h>

#define N_NODES 50000
#define N_EDGES 800000
#define IN_DIM  384
#define HID     128
#define HEADS   4
#define OUT_DIM 5
#define F1      (HEADS * HID)   /* 512 */
#define NEG_SLOPE 0.2f
#define EPS_F   1e-16f

typedef __attribute__((ext_vector_type(16))) __bf16 v16bf;
typedef __attribute__((ext_vector_type(8)))  float  v8f;

// ---------- helpers ----------
__device__ __forceinline__ unsigned short f2bf(float f) {
    unsigned u = __float_as_uint(f);
    unsigned r = u + 0x7FFFu + ((u >> 16) & 1u);   // round-to-nearest-even
    return (unsigned short)(r >> 16);
}
// monotonic float<->uint mapping so atomicMax(u32) == float max
__device__ __forceinline__ unsigned encf(float f) {
    unsigned u = __float_as_uint(f);
    return (u & 0x80000000u) ? ~u : (u | 0x80000000u);
}
__device__ __forceinline__ float decf(unsigned u) {
    return (u & 0x80000000u) ? __uint_as_float(u ^ 0x80000000u)
                             : __uint_as_float(~u);
}
__device__ __forceinline__ float lrelu(float v) { return v > 0.f ? v : v * NEG_SLOPE; }

// ---------- init ----------
__global__ void fill_f32(float* p, float v, int n) {
    int i = blockIdx.x * blockDim.x + threadIdx.x;
    if (i < n) p[i] = v;
}
__global__ void fill_u32(unsigned* p, unsigned v, int n) {
    int i = blockIdx.x * blockDim.x + threadIdx.x;
    if (i < n) p[i] = v;
}

// ---------- conversion: X -> bf16, W1 -> bf16 transposed [512,384] ----------
__global__ void cvt_x(const float* __restrict__ x, unsigned short* __restrict__ xb, int n) {
    int i = blockIdx.x * blockDim.x + threadIdx.x;
    if (i < n) xb[i] = f2bf(x[i]);
}
__global__ void cvt_w1t(const float* __restrict__ w1, unsigned short* __restrict__ w1t) {
    int i = blockIdx.x * blockDim.x + threadIdx.x;   // over 512*384
    if (i >= F1 * IN_DIM) return;
    int n = i / IN_DIM, k = i - n * IN_DIM;
    w1t[i] = f2bf(w1[(size_t)k * F1 + n]);
}

// ---------- GEMM1: h1[50000,512] = Xb[50000,384] @ W1 (bf16 WMMA, f32 acc) ----------
// grid (3125, 2), block 128 (4 waves). Wave -> one 16x64 output strip.
__global__ void __launch_bounds__(128)
gemm1_wmma(const unsigned short* __restrict__ Xb,
           const unsigned short* __restrict__ W1T,
           float* __restrict__ h1) {
    const int wave = threadIdx.x >> 5;
    const int lane = threadIdx.x & 31;
    const int half = lane >> 4;          // 0: lanes 0-15, 1: lanes 16-31
    const int lm   = lane & 15;
    const int mTile   = blockIdx.x;                     // 0..3124
    const int nGroup  = blockIdx.y * 4 + wave;          // 0..7
    const int colBase = nGroup * 64;

    const int row = mTile * 16 + lm;
    const unsigned short* arow = Xb + (size_t)row * IN_DIM;

    v8f acc0 = {0,0,0,0,0,0,0,0};
    v8f acc[4] = {acc0, acc0, acc0, acc0};

    for (int kt = 0; kt < IN_DIM / 32; ++kt) {
        const int K0 = kt * 32;
        // A fragment: lane<16 holds K {K0..K0+7, K0+16..K0+23}; lane>=16 shifts by 8
        union { uint32_t u[8]; v16bf v; } A;
        {
            const uint4 a0 = *(const uint4*)(arow + K0 + half * 8);
            const uint4 a1 = *(const uint4*)(arow + K0 + half * 8 + 16);
            A.u[0]=a0.x; A.u[1]=a0.y; A.u[2]=a0.z; A.u[3]=a0.w;
            A.u[4]=a1.x; A.u[5]=a1.y; A.u[6]=a1.z; A.u[7]=a1.w;
        }
        #pragma unroll
        for (int nt = 0; nt < 4; ++nt) {
            const int col = colBase + nt * 16 + lm;
            const unsigned short* bcol = W1T + (size_t)col * IN_DIM + K0 + half * 8;
            union { uint32_t u[8]; v16bf v; } B;
            const uint4 b0 = *(const uint4*)(bcol);
            const uint4 b1 = *(const uint4*)(bcol + 16);
            B.u[0]=b0.x; B.u[1]=b0.y; B.u[2]=b0.z; B.u[3]=b0.w;
            B.u[4]=b1.x; B.u[5]=b1.y; B.u[6]=b1.z; B.u[7]=b1.w;
            acc[nt] = __builtin_amdgcn_wmma_f32_16x16x32_bf16(
                false, A.v, false, B.v, (short)0, acc[nt], false, false);
        }
    }
    #pragma unroll
    for (int nt = 0; nt < 4; ++nt) {
        const int col = colBase + nt * 16 + lm;
        float* orow = h1 + (size_t)(mTile * 16 + half * 8) * F1 + col;
        #pragma unroll
        for (int g = 0; g < 8; ++g) orow[(size_t)g * F1] = acc[nt][g];
    }
}

// ---------- attention coefficients, layer 1 ----------
__global__ void att1(const float* __restrict__ h1, const float* __restrict__ asr,
                     const float* __restrict__ ads, float* __restrict__ als,
                     float* __restrict__ ald) {
    int i = blockIdx.x * blockDim.x + threadIdx.x;   // N*4
    if (i >= N_NODES * HEADS) return;
    int n = i >> 2, h = i & 3;
    const float* hp = h1 + (size_t)n * F1 + h * HID;
    const float* sp = asr + h * HID;
    const float* dp = ads + h * HID;
    float s = 0.f, d = 0.f;
    for (int k = 0; k < HID; ++k) { float v = hp[k]; s += v * sp[k]; d += v * dp[k]; }
    als[i] = s; ald[i] = d;
}

// ---------- edge softmax passes, layer 1 ----------
__global__ void edge_max1(const int* __restrict__ ei, const float* __restrict__ als,
                          const float* __restrict__ ald, unsigned* __restrict__ m1) {
    int i = blockIdx.x * blockDim.x + threadIdx.x;   // E*4
    if (i >= N_EDGES * HEADS) return;
    int e = i >> 2, h = i & 3;
    int s = ei[e], d = ei[N_EDGES + e];
    float v = lrelu(als[s * 4 + h] + ald[d * 4 + h]);
    atomicMax(&m1[d * 4 + h], encf(v));
}
__global__ void edge_exp1(const int* __restrict__ ei, const float* __restrict__ als,
                          const float* __restrict__ ald, const unsigned* __restrict__ m1,
                          float* __restrict__ ex1, float* __restrict__ den1) {
    int i = blockIdx.x * blockDim.x + threadIdx.x;
    if (i >= N_EDGES * HEADS) return;
    int e = i >> 2, h = i & 3;
    int s = ei[e], d = ei[N_EDGES + e];
    float v = lrelu(als[s * 4 + h] + ald[d * 4 + h]);
    float ex = __expf(v - decf(m1[d * 4 + h]));
    ex1[i] = ex;
    atomicAdd(&den1[d * 4 + h], ex);
}
// one wave per edge; lanes stride the 512 channels (coalesced gather + L2 atomics)
__global__ void __launch_bounds__(256)
edge_scatter1(const int* __restrict__ ei, const float* __restrict__ ex1,
              const float* __restrict__ den1, const float* __restrict__ h1,
              float* __restrict__ g1) {
    int e = blockIdx.x * 8 + (threadIdx.x >> 5);
    int lane = threadIdx.x & 31;
    int s = ei[e], d = ei[N_EDGES + e];
    float coef[HEADS];
    #pragma unroll
    for (int h = 0; h < HEADS; ++h)
        coef[h] = ex1[(size_t)e * 4 + h] / (den1[d * 4 + h] + EPS_F);
    const float* hs = h1 + (size_t)s * F1;
    float* go = g1 + (size_t)d * F1;
    #pragma unroll
    for (int it = 0; it < F1 / 32; ++it) {
        int c = lane + it * 32;
        atomicAdd(&go[c], hs[c] * coef[c >> 7]);
    }
}

// ---------- bias + ELU (layer 1 output -> h2, in place) ----------
__global__ void elu_bias(float* __restrict__ g1, const float* __restrict__ b1) {
    int i = blockIdx.x * blockDim.x + threadIdx.x;
    if (i >= N_NODES * F1) return;
    float v = g1[i] + b1[i & (F1 - 1)];
    g1[i] = v > 0.f ? v : expm1f(v);
}

// ---------- layer 2 ----------
__global__ void gemm2(const float* __restrict__ g1, const float* __restrict__ W2,
                      float* __restrict__ z) {
    int i = blockIdx.x * blockDim.x + threadIdx.x;   // N*5
    if (i >= N_NODES * OUT_DIM) return;
    int n = i / OUT_DIM, c = i - n * OUT_DIM;
    const float* r = g1 + (size_t)n * F1;
    float acc = 0.f;
    for (int k = 0; k < F1; ++k) acc += r[k] * W2[k * OUT_DIM + c];
    z[i] = acc;
}
__global__ void att2(const float* __restrict__ z, const float* __restrict__ asr,
                     const float* __restrict__ ads, float* __restrict__ als,
                     float* __restrict__ ald) {
    int n = blockIdx.x * blockDim.x + threadIdx.x;
    if (n >= N_NODES) return;
    float s = 0.f, d = 0.f;
    #pragma unroll
    for (int c = 0; c < OUT_DIM; ++c) { float v = z[n * OUT_DIM + c]; s += v * asr[c]; d += v * ads[c]; }
    als[n] = s; ald[n] = d;
}
__global__ void edge_max2(const int* __restrict__ ei, const float* __restrict__ als,
                          const float* __restrict__ ald, unsigned* __restrict__ m2) {
    int e = blockIdx.x * blockDim.x + threadIdx.x;
    if (e >= N_EDGES) return;
    int s = ei[e], d = ei[N_EDGES + e];
    atomicMax(&m2[d], encf(lrelu(als[s] + ald[d])));
}
__global__ void edge_exp2(const int* __restrict__ ei, const float* __restrict__ als,
                          const float* __restrict__ ald, const unsigned* __restrict__ m2,
                          float* __restrict__ ex2, float* __restrict__ den2) {
    int e = blockIdx.x * blockDim.x + threadIdx.x;
    if (e >= N_EDGES) return;
    int s = ei[e], d = ei[N_EDGES + e];
    float ex = __expf(lrelu(als[s] + ald[d]) - decf(m2[d]));
    ex2[e] = ex;
    atomicAdd(&den2[d], ex);
}
__global__ void edge_scatter2(const int* __restrict__ ei, const float* __restrict__ ex2,
                              const float* __restrict__ den2, const float* __restrict__ z,
                              float* __restrict__ out) {
    int e = blockIdx.x * blockDim.x + threadIdx.x;
    if (e >= N_EDGES) return;
    int s = ei[e], d = ei[N_EDGES + e];
    float coef = ex2[e] / (den2[d] + EPS_F);
    #pragma unroll
    for (int c = 0; c < OUT_DIM; ++c)
        atomicAdd(&out[(size_t)d * OUT_DIM + c], z[(size_t)s * OUT_DIM + c] * coef);
}
__global__ void bias2(float* __restrict__ out, const float* __restrict__ b2) {
    int i = blockIdx.x * blockDim.x + threadIdx.x;
    if (i >= N_NODES * OUT_DIM) return;
    out[i] += b2[i % OUT_DIM];
}

// ---------- host ----------
extern "C" void kernel_launch(void* const* d_in, const int* in_sizes, int n_in,
                              void* d_out, int out_size, void* d_ws, size_t ws_size,
                              hipStream_t stream) {
    const float* x   = (const float*)d_in[0];
    const int*   ei  = (const int*)  d_in[1];
    const float* W1  = (const float*)d_in[2];
    const float* as1 = (const float*)d_in[3];
    const float* ad1 = (const float*)d_in[4];
    const float* b1  = (const float*)d_in[5];
    const float* W2  = (const float*)d_in[6];
    const float* as2 = (const float*)d_in[7];
    const float* ad2 = (const float*)d_in[8];
    const float* b2  = (const float*)d_in[9];
    float* out = (float*)d_out;

    char* w = (char*)d_ws;
    auto carve = [&](size_t bytes) -> void* {
        void* p = (void*)w;
        w += (bytes + 255) & ~(size_t)255;
        return p;
    };
    unsigned short* Xb   = (unsigned short*)carve((size_t)N_NODES * IN_DIM * 2);
    unsigned short* W1T  = (unsigned short*)carve((size_t)F1 * IN_DIM * 2);
    float*    h1   = (float*)   carve((size_t)N_NODES * F1 * 4);
    float*    g1   = (float*)   carve((size_t)N_NODES * F1 * 4);
    float*    als1 = (float*)   carve((size_t)N_NODES * HEADS * 4);
    float*    ald1 = (float*)   carve((size_t)N_NODES * HEADS * 4);
    unsigned* m1   = (unsigned*)carve((size_t)N_NODES * HEADS * 4);
    float*    den1 = (float*)   carve((size_t)N_NODES * HEADS * 4);
    float*    ex1  = (float*)   carve((size_t)N_EDGES * HEADS * 4);
    float*    z    = (float*)   carve((size_t)N_NODES * OUT_DIM * 4);
    float*    als2 = (float*)   carve((size_t)N_NODES * 4);
    float*    ald2 = (float*)   carve((size_t)N_NODES * 4);
    unsigned* m2   = (unsigned*)carve((size_t)N_NODES * 4);
    float*    den2 = (float*)   carve((size_t)N_NODES * 4);
    float*    ex2  = (float*)   carve((size_t)N_EDGES * 4);

    const int T = 256;
    const unsigned ENC_NEG_INF = 0x007FFFFFu;   // encf(-inf)
    auto nb = [](int n, int t) { return (n + t - 1) / t; };

    // zero / init accumulators (fresh every call)
    fill_f32<<<nb(N_NODES * F1, T), T, 0, stream>>>(g1, 0.f, N_NODES * F1);
    fill_f32<<<nb(N_NODES * OUT_DIM, T), T, 0, stream>>>(out, 0.f, N_NODES * OUT_DIM);
    fill_f32<<<nb(N_NODES * HEADS, T), T, 0, stream>>>(den1, 0.f, N_NODES * HEADS);
    fill_f32<<<nb(N_NODES, T), T, 0, stream>>>(den2, 0.f, N_NODES);
    fill_u32<<<nb(N_NODES * HEADS, T), T, 0, stream>>>(m1, ENC_NEG_INF, N_NODES * HEADS);
    fill_u32<<<nb(N_NODES, T), T, 0, stream>>>(m2, ENC_NEG_INF, N_NODES);

    // bf16 staging for WMMA GEMM
    cvt_x  <<<nb(N_NODES * IN_DIM, T), T, 0, stream>>>(x, Xb, N_NODES * IN_DIM);
    cvt_w1t<<<nb(F1 * IN_DIM, T), T, 0, stream>>>(W1, W1T);

    // layer 1
    gemm1_wmma<<<dim3(N_NODES / 16, 2), 128, 0, stream>>>(Xb, W1T, h1);
    att1<<<nb(N_NODES * HEADS, T), T, 0, stream>>>(h1, as1, ad1, als1, ald1);
    edge_max1<<<nb(N_EDGES * HEADS, T), T, 0, stream>>>(ei, als1, ald1, m1);
    edge_exp1<<<nb(N_EDGES * HEADS, T), T, 0, stream>>>(ei, als1, ald1, m1, ex1, den1);
    edge_scatter1<<<N_EDGES / 8, 256, 0, stream>>>(ei, ex1, den1, h1, g1);
    elu_bias<<<nb(N_NODES * F1, T), T, 0, stream>>>(g1, b1);

    // layer 2
    gemm2<<<nb(N_NODES * OUT_DIM, T), T, 0, stream>>>(g1, W2, z);
    att2<<<nb(N_NODES, T), T, 0, stream>>>(z, as2, ad2, als2, ald2);
    edge_max2<<<nb(N_EDGES, T), T, 0, stream>>>(ei, als2, ald2, m2);
    edge_exp2<<<nb(N_EDGES, T), T, 0, stream>>>(ei, als2, ald2, m2, ex2, den2);
    edge_scatter2<<<nb(N_EDGES, T), T, 0, stream>>>(ei, ex2, den2, z, out);
    bias2<<<nb(N_NODES * OUT_DIM, T), T, 0, stream>>>(out, b2);
}
